// DIVIDE_11622181503659
// MI455X (gfx1250) — matrix operand order repaired and checked by reading, not compiled, not run on hardware
//
#include <hip/hip_runtime.h>

// ---------------------------------------------------------------------------
// CDNA5 (gfx1250, wave32) types
// ---------------------------------------------------------------------------
typedef __attribute__((ext_vector_type(16))) __bf16 v16bf;
typedef __attribute__((ext_vector_type(8)))  float  v8f;
typedef __attribute__((ext_vector_type(4)))  int    v4i;

struct alignas(16) U4 { unsigned int x, y, z, w; };
union Frag { U4 q[2]; v16bf v; };

__device__ __forceinline__ unsigned short f2bf_bits(float f) {
  union { float f; unsigned u; } a; a.f = f;
  unsigned r = a.u + 0x7FFFu + ((a.u >> 16) & 1u);   // round-to-nearest-even
  return (unsigned short)(r >> 16);
}

// Async global->LDS staging (CDNA5 ASYNCcnt path), with portable fallback.
#if defined(__HIP_DEVICE_COMPILE__)
# if __has_builtin(__builtin_amdgcn_global_load_async_to_lds_b128) && \
     __has_builtin(__builtin_amdgcn_s_wait_asynccnt)
#  define USE_ASYNC_LDS 1
# endif
#endif

#define GAS __attribute__((address_space(1)))
#define LAS __attribute__((address_space(3)))

// ---------------------------------------------------------------------------
// All-NT bf16 WMMA GEMM:
//   C[M,Nc] = A[M,K] * B[Nc,K]^T   (A, B bf16 row-major, K contiguous in both)
// Block tile 128x128, BK=32, 256 threads = 8 waves (wave32), wave tile 32x64.
// Tiles staged to LDS with zero VALU: one b128 per lane per async op.
// M,Nc multiples of 128; K multiple of 32 (true for all shapes here).
// ---------------------------------------------------------------------------
#define ASTR 40   // LDS row stride in bf16 elements (80B, multiple of 16B)

template<bool BIAS, bool RELU, bool BF16OUT>
__global__ __launch_bounds__(256) void gemm_bf16_nt(
    const unsigned short* __restrict__ A, const unsigned short* __restrict__ B,
    const float* __restrict__ bias, float* __restrict__ Cf,
    unsigned short* __restrict__ Cb, int M, int Nc, int K,
    int lda, int ldb, int ldc)
{
  __shared__ unsigned short As[128 * ASTR];  // [m][k]
  __shared__ unsigned short Bs[128 * ASTR];  // [n][k]

  const int t    = threadIdx.x;
  const int wave = t >> 5;
  const int lane = t & 31;
  const int half = lane >> 4;
  const int lnn  = lane & 15;
  const int wm   = wave >> 1;     // 0..3 -> M offset wm*32
  const int wn   = wave & 1;      // 0..1 -> N offset wn*64
  const int m0   = blockIdx.y * 128;
  const int n0   = blockIdx.x * 128;

  v8f acc[2][4] = {};

  for (int k0 = 0; k0 < K; k0 += 32) {
    // ---- stage A (128x32) and B (128x32) bf16 tiles into LDS ----
#if defined(USE_ASYNC_LDS)
#pragma unroll
    for (int i = 0; i < 2; ++i) {
      int flat = i * 256 + t;            // 512 x b128 per matrix tile
      int r  = flat >> 2;
      int c8 = (flat & 3) << 3;          // bf16 element offset within row
      __builtin_amdgcn_global_load_async_to_lds_b128(
          (GAS v4i*)(A + (size_t)(m0 + r) * lda + k0 + c8),
          (LAS v4i*)(&As[r * ASTR + c8]), 0, 0);
      __builtin_amdgcn_global_load_async_to_lds_b128(
          (GAS v4i*)(B + (size_t)(n0 + r) * ldb + k0 + c8),
          (LAS v4i*)(&Bs[r * ASTR + c8]), 0, 0);
    }
    __builtin_amdgcn_s_wait_asynccnt(0);
#else
#pragma unroll
    for (int i = 0; i < 2; ++i) {
      int flat = i * 256 + t;
      int r  = flat >> 2;
      int c8 = (flat & 3) << 3;
      *(U4*)&As[r * ASTR + c8] = *(const U4*)(A + (size_t)(m0 + r) * lda + k0 + c8);
      *(U4*)&Bs[r * ASTR + c8] = *(const U4*)(B + (size_t)(n0 + r) * ldb + k0 + c8);
    }
#endif
    __syncthreads();

    // ---- fragments per CDNA5 16-bit WMMA VGPR layouts ----
    // A 16x32: lane m=lnn, elem idx: K = (idx<8 ? 0 : 16) + half*8 + (idx&7)
    Frag fa[2];
#pragma unroll
    for (int r = 0; r < 2; ++r) {
      const unsigned short* pa = &As[(wm * 32 + r * 16 + lnn) * ASTR + half * 8];
      fa[r].q[0] = *(const U4*)pa;
      fa[r].q[1] = *(const U4*)(pa + 16);
    }
    // B 32x16: lane n=lnn, elem idx: K = (idx<8 ? 0 : 16) + half*8 + (idx&7)
    Frag fb[4];
#pragma unroll
    for (int c = 0; c < 4; ++c) {
      const unsigned short* pb = &Bs[(wn * 64 + c * 16 + lnn) * ASTR + half * 8];
      fb[c].q[0] = *(const U4*)pb;
      fb[c].q[1] = *(const U4*)(pb + 16);
    }
#pragma unroll
    for (int r = 0; r < 2; ++r)
#pragma unroll
      for (int c = 0; c < 4; ++c)
        acc[r][c] = __builtin_amdgcn_wmma_f32_16x16x32_bf16(
            false, fa[r].v, false, fb[c].v, (short)0, acc[r][c], false, false);
    __syncthreads();
  }

  // ---- epilogue: C layout elem e -> M = e + 8*half, N = lnn ----
#pragma unroll
  for (int r = 0; r < 2; ++r) {
#pragma unroll
    for (int c = 0; c < 4; ++c) {
      const int rowb = m0 + wm * 32 + r * 16 + 8 * half;
      const int col  = n0 + wn * 64 + c * 16 + lnn;
      float bv = 0.f;
      if (BIAS) bv = bias[col];
#pragma unroll
      for (int e = 0; e < 8; ++e) {
        float v = acc[r][c][e] + bv;
        if (RELU) v = fmaxf(v, 0.f);
        if (BF16OUT) Cb[(size_t)(rowb + e) * ldc + col] = f2bf_bits(v);
        else         Cf[(size_t)(rowb + e) * ldc + col] = v;
      }
    }
  }
}

// ---------------------------------------------------------------------------
// Data preparation kernels
// ---------------------------------------------------------------------------
__global__ void zero_f32(float* __restrict__ p, int n) {
  int i = blockIdx.x * 256 + threadIdx.x;
  if (i < n) p[i] = 0.f;
}

// straight f32 -> bf16 convert
__global__ void cvt_bf16(const float* __restrict__ in, unsigned short* __restrict__ out, int n) {
  int i = blockIdx.x * 256 + threadIdx.x;
  if (i < n) out[i] = f2bf_bits(in[i]);
}

// EMA for f32 vectors (BN affine params)
__global__ void ema_kernel(float* __restrict__ dst, const float* __restrict__ tg,
                           const float* __restrict__ w, const float* __restrict__ mom, int n) {
  int i = blockIdx.x * 256 + threadIdx.x;
  if (i < n) {
    float m = *mom;
    dst[i] = tg[i] * m + w[i] * (1.f - m);
  }
}

// [R,C] f32 (optionally EMA of two inputs) -> [C,R] bf16, tiled via LDS.
template<bool EMA>
__global__ __launch_bounds__(256) void transpose_cvt(
    const float* __restrict__ Wsrc, const float* __restrict__ Tsrc,
    const float* __restrict__ mom, unsigned short* __restrict__ out, int R, int C) {
  __shared__ float tile[32][33];
  const int rb = blockIdx.y * 32, cb = blockIdx.x * 32;
  const int tx = threadIdx.x & 31, ty = threadIdx.x >> 5;   // 8 row-groups
  for (int i = ty; i < 32; i += 8) {
    float v = Wsrc[(size_t)(rb + i) * C + cb + tx];
    if (EMA) {
      float m = *mom;
      v = Tsrc[(size_t)(rb + i) * C + cb + tx] * m + v * (1.f - m);
    }
    tile[i][tx] = v;
  }
  __syncthreads();
  for (int i = ty; i < 32; i += 8)
    out[(size_t)(cb + i) * R + rb + tx] = f2bf_bits(tile[tx][i]);
}

// ---------------------------------------------------------------------------
// BatchNorm (training mode, biased var)
// ---------------------------------------------------------------------------
__global__ void bn_partial(const float* __restrict__ X, float* __restrict__ stats,
                           int rows, int cols) {
  int c  = blockIdx.x * 256 + threadIdx.x;
  int r0 = blockIdx.y * 256;
  float s = 0.f, s2 = 0.f;
  for (int r = r0; r < r0 + 256; ++r) {
    float v = X[(size_t)r * cols + c];
    s += v; s2 += v * v;
  }
  atomicAdd(&stats[c], s);
  atomicAdd(&stats[cols + c], s2);
}

// Writes bf16 output always (next GEMM input); f32 copy if F32OUT.
template<bool AFFINE, bool RELU, bool F32OUT>
__global__ void bn_apply(const float* __restrict__ X, const float* __restrict__ stats,
                         const float* __restrict__ g, const float* __restrict__ b,
                         float* __restrict__ Yf, unsigned short* __restrict__ Yb,
                         int rows, int cols) {
  size_t i = (size_t)blockIdx.x * 256 + threadIdx.x;
  size_t total = (size_t)rows * cols;
  if (i >= total) return;
  int c = (int)(i % cols);
  float inv_n = 1.f / (float)rows;
  float mean = stats[c] * inv_n;
  float var  = stats[cols + c] * inv_n - mean * mean;
  float inv  = rsqrtf(var + 1e-5f);
  float v = (X[i] - mean) * inv;
  if (AFFINE) v = v * g[c] + b[c];
  if (RELU)   v = fmaxf(v, 0.f);
  Yb[i] = f2bf_bits(v);
  if (F32OUT) Yf[i] = v;
}

// ---------------------------------------------------------------------------
// Row-wise kernels (l2norm, kernel_affinity, contrastive)
// ---------------------------------------------------------------------------
__global__ void l2norm_rows(float* __restrict__ X, int F) {
  __shared__ float red[256];
  float* p = X + (size_t)blockIdx.x * F;
  int tid = threadIdx.x;
  float s = 0.f;
  for (int i = tid; i < F; i += 256) { float v = p[i]; s += v * v; }
  red[tid] = s; __syncthreads();
  for (int off = 128; off > 0; off >>= 1) {
    if (tid < off) red[tid] += red[tid + off];
    __syncthreads();
  }
  float inv = 1.f / fmaxf(sqrtf(red[0]), 1e-12f);
  for (int i = tid; i < F; i += 256) p[i] *= inv;
}

__global__ void affinity_rows(const float* __restrict__ G, float* __restrict__ MP, int n) {
  __shared__ float s1[256];
  __shared__ float s2[256];
  __shared__ int   si[256];
  const int i = blockIdx.x, tid = threadIdx.x;
  const float* g = G + (size_t)i * n;
  float* mp = MP + (size_t)i * n;

  // pass 1: S = sum_j exp(-clip(2-2g,0)/0.1)
  float s = 0.f;
  for (int j = tid; j < n; j += 256) {
    float d = fmaxf(2.f - 2.f * g[j], 0.f);
    s += expf(-d * 10.f);
  }
  s1[tid] = s; __syncthreads();
  for (int off = 128; off > 0; off >>= 1) {
    if (tid < off) s1[tid] += s1[tid + off];
    __syncthreads();
  }
  const float invS = 1.f / s1[0];
  __syncthreads();

  const float ai = expf(-fmaxf(2.f - 2.f * g[i], 0.f) * 10.f) * invS;  // diag(A)_i

  // pass 2: argmin_j(diff + eye), first-occurrence ties; sumE = sum_{j!=i} exp(A_j)
  float best = 1e30f; int bidx = 0x7fffffff; float se = 0.f;
  for (int j = tid; j < n; j += 256) {
    float Aj = expf(-fmaxf(2.f - 2.f * g[j], 0.f) * 10.f) * invS;
    float dm = fabsf(ai - Aj) + (j == i ? 1.f : 0.f);
    if (dm < best || (dm == best && j < bidx)) { best = dm; bidx = j; }
    if (j != i) se += expf(Aj);
  }
  s1[tid] = se; s2[tid] = best; si[tid] = bidx; __syncthreads();
  for (int off = 128; off > 0; off >>= 1) {
    if (tid < off) {
      s1[tid] += s1[tid + off];
      float v2 = s2[tid + off]; int i2 = si[tid + off];
      if (v2 < s2[tid] || (v2 == s2[tid] && i2 < si[tid])) { s2[tid] = v2; si[tid] = i2; }
    }
    __syncthreads();
  }
  const float invSE = 1.f / s1[0];
  const int idx = si[0];
  __syncthreads();

  // pass 3: mp = A * dyn
  for (int j = tid; j < n; j += 256) {
    float Aj   = expf(-fmaxf(2.f - 2.f * g[j], 0.f) * 10.f) * invS;
    float diff = fabsf(ai - Aj);
    float thr  = (diff < 0.8f) ? 1.f : 0.f;
    float oh   = (j == idx) ? 1.f : 0.f;
    float eye  = (j == i) ? 1.f : 0.f;
    float tkt  = thr * (oh + eye);
    float E    = expf(Aj) * (1.f - eye);
    float wgt  = 1.f - E * invSE;
    float dyn  = tkt * wgt + (1.f - tkt);
    mp[j] = Aj * dyn;
  }
}

// Row contribution: lse - (sum_j t*mp)/(sum_j mp), t = 2*G (temp 0.5).
__global__ void loss_rows(const float* __restrict__ G, const float* __restrict__ MP,
                          float* __restrict__ loss, float weight, int n) {
  __shared__ float s1[256];
  __shared__ float s2[256];
  __shared__ float s3[256];
  const int i = blockIdx.x, tid = threadIdx.x;
  const float* g  = G  + (size_t)i * n;
  const float* mp = MP + (size_t)i * n;

  float mx = -3.4e38f;
  for (int j = tid; j < n; j += 256) mx = fmaxf(mx, 2.f * g[j]);
  s1[tid] = mx; __syncthreads();
  for (int off = 128; off > 0; off >>= 1) {
    if (tid < off) s1[tid] = fmaxf(s1[tid], s1[tid + off]);
    __syncthreads();
  }
  mx = s1[0]; __syncthreads();

  float se = 0.f, smp = 0.f, stm = 0.f;
  for (int j = tid; j < n; j += 256) {
    float tv = 2.f * g[j];
    se  += expf(tv - mx);
    float m = mp[j];
    smp += m;
    stm += tv * m;
  }
  s1[tid] = se; s2[tid] = smp; s3[tid] = stm; __syncthreads();
  for (int off = 128; off > 0; off >>= 1) {
    if (tid < off) { s1[tid] += s1[tid + off]; s2[tid] += s2[tid + off]; s3[tid] += s3[tid + off]; }
    __syncthreads();
  }
  if (tid == 0) {
    float lse = mx + logf(s1[0]);
    atomicAdd(loss, weight * (lse - s3[0] / s2[0]));
  }
}

// ---------------------------------------------------------------------------
// Host orchestration
// ---------------------------------------------------------------------------
namespace {
constexpr int V = 3, N = 4096, D = 2048, H = 2048, F = 512;
typedef unsigned short bf16_t;

void run_gemm(const bf16_t* A, const bf16_t* B, float* C,
              int M, int Nc, int K, hipStream_t s) {
  gemm_bf16_nt<false, false, false><<<dim3(Nc / 128, M / 128), 256, 0, s>>>(
      A, B, nullptr, C, nullptr, M, Nc, K, K, K, Nc);
}
void run_gemm_bias_relu_bf16(const bf16_t* A, const bf16_t* B, const float* bias,
                             bf16_t* C, int M, int Nc, int K, hipStream_t s) {
  gemm_bf16_nt<true, true, true><<<dim3(Nc / 128, M / 128), 256, 0, s>>>(
      A, B, bias, nullptr, C, M, Nc, K, K, K, Nc);
}
void run_gemm_bias(const bf16_t* A, const bf16_t* B, const float* bias, float* C,
                   int M, int Nc, int K, hipStream_t s) {
  gemm_bf16_nt<true, false, false><<<dim3(Nc / 128, M / 128), 256, 0, s>>>(
      A, B, bias, C, nullptr, M, Nc, K, K, K, Nc);
}
// BN: stats then apply; hidden layers write bf16 only; z layer writes f32 + bf16.
void run_bn_hidden(const float* X, const float* g, const float* b, bf16_t* Yb,
                   float* stats, int rows, int cols, hipStream_t s) {
  zero_f32<<<(2 * cols + 255) / 256, 256, 0, s>>>(stats, 2 * cols);
  bn_partial<<<dim3(cols / 256, rows / 256), 256, 0, s>>>(X, stats, rows, cols);
  dim3 gr((unsigned)(((size_t)rows * cols + 255) / 256));
  bn_apply<true, true, false><<<gr, 256, 0, s>>>(X, stats, g, b, nullptr, Yb, rows, cols);
}
void run_bn_final(const float* X, float* Yf, bf16_t* Yb, float* stats,
                  int rows, int cols, hipStream_t s) {
  zero_f32<<<(2 * cols + 255) / 256, 256, 0, s>>>(stats, 2 * cols);
  bn_partial<<<dim3(cols / 256, rows / 256), 256, 0, s>>>(X, stats, rows, cols);
  dim3 gr((unsigned)(((size_t)rows * cols + 255) / 256));
  bn_apply<false, false, true><<<gr, 256, 0, s>>>(X, stats, nullptr, nullptr, Yf, Yb, rows, cols);
}
void run_transpose(const float* W, bf16_t* out, int R, int C, hipStream_t s) {
  transpose_cvt<false><<<dim3(C / 32, R / 32), 256, 0, s>>>(W, nullptr, nullptr, out, R, C);
}
void run_transpose_ema(const float* W, const float* T, const float* mom, bf16_t* out,
                       int R, int C, hipStream_t s) {
  transpose_cvt<true><<<dim3(C / 32, R / 32), 256, 0, s>>>(W, T, mom, out, R, C);
}
}  // namespace

extern "C" void kernel_launch(void* const* d_in, const int* in_sizes, int n_in,
                              void* d_out, int out_size, void* d_ws, size_t ws_size,
                              hipStream_t stream) {
  (void)in_sizes; (void)n_in; (void)out_size; (void)ws_size;
  const float* x   = (const float*)d_in[0];
  const float* W1  = (const float*)d_in[1];
  const float* g1  = (const float*)d_in[2];
  const float* b1  = (const float*)d_in[3];
  const float* W2  = (const float*)d_in[4];
  const float* g2  = (const float*)d_in[5];
  const float* b2  = (const float*)d_in[6];
  const float* W3  = (const float*)d_in[7];
  const float* tW1 = (const float*)d_in[8];
  const float* tg1 = (const float*)d_in[9];
  const float* tb1 = (const float*)d_in[10];
  const float* tW2 = (const float*)d_in[11];
  const float* tg2 = (const float*)d_in[12];
  const float* tb2 = (const float*)d_in[13];
  const float* tW3 = (const float*)d_in[14];
  const float* dW1 = (const float*)d_in[15];
  const float* db1 = (const float*)d_in[16];
  const float* dW2 = (const float*)d_in[17];
  const float* db2 = (const float*)d_in[18];
  const float* mom = (const float*)d_in[19];
  float* out = (float*)d_out;

  // workspace carve (256B-aligned bump allocator)
  char* wp = (char*)d_ws;
  auto alloc = [&](size_t bytes) -> void* {
    void* p = wp;
    wp += (bytes + 255) & ~(size_t)255;
    return p;
  };
  // bf16 matrices (weights transposed to [N,K])
  bf16_t* xb   = (bf16_t*)alloc((size_t)N * D * 2);
  bf16_t* w1T  = (bf16_t*)alloc((size_t)H * D * 2);
  bf16_t* w2T  = (bf16_t*)alloc((size_t)H * H * 2);
  bf16_t* w3T  = (bf16_t*)alloc((size_t)F * H * 2);
  bf16_t* e1T  = (bf16_t*)alloc((size_t)H * D * 2);
  bf16_t* e2T  = (bf16_t*)alloc((size_t)H * H * 2);
  bf16_t* e3T  = (bf16_t*)alloc((size_t)F * H * 2);
  bf16_t* d1T  = (bf16_t*)alloc((size_t)H * F * 2);
  bf16_t* d2T  = (bf16_t*)alloc((size_t)F * H * 2);
  bf16_t* abuf = (bf16_t*)alloc((size_t)N * H * 2);   // BN bf16 activations
  bf16_t* mlpb = (bf16_t*)alloc((size_t)N * H * 2);   // mlp hidden bf16
  bf16_t* zqb  = (bf16_t*)alloc((size_t)V * N * F * 2);
  bf16_t* zkb  = (bf16_t*)alloc((size_t)V * N * F * 2);
  bf16_t* ppb  = (bf16_t*)alloc((size_t)V * N * F * 2);
  // f32 buffers
  float* eg1  = (float*)alloc(H * 4);
  float* eb1  = (float*)alloc(H * 4);
  float* eg2  = (float*)alloc(H * 4);
  float* eb2  = (float*)alloc(H * 4);
  float* buf1 = (float*)alloc((size_t)N * H * 4);     // GEMM f32 out / BN in
  float* zq   = (float*)alloc((size_t)V * N * F * 4);
  float* zk   = (float*)alloc((size_t)V * N * F * 4);
  float* pp   = (float*)alloc((size_t)V * N * F * 4);
  float* Gb   = (float*)alloc((size_t)N * N * 4);
  float* mpb  = (float*)alloc((size_t)N * N * 4);
  float* stats = (float*)alloc(2 * H * 4);

  zero_f32<<<1, 256, 0, stream>>>(out, 1);

  for (int v = 0; v < V; ++v) {
    const size_t oW1 = (size_t)v * D * H;
    const size_t oW3 = (size_t)v * H * F;
    const size_t oV  = (size_t)v * H;
    const size_t oFv = (size_t)v * F;
    const size_t oX  = (size_t)v * N * D;
    const size_t oZ  = (size_t)v * N * F;
    const size_t odW1 = (size_t)v * F * H;

    // one-time conversions / transposes for this view
    cvt_bf16<<<(N * D + 255) / 256, 256, 0, stream>>>(x + oX, xb, N * D);
    run_transpose(W1 + oW1, w1T, D, H, stream);
    run_transpose(W2 + oW1, w2T, H, H, stream);
    run_transpose(W3 + oW3, w3T, H, F, stream);
    run_transpose_ema(W1 + oW1, tW1 + oW1, mom, e1T, D, H, stream);
    run_transpose_ema(W2 + oW1, tW2 + oW1, mom, e2T, H, H, stream);
    run_transpose_ema(W3 + oW3, tW3 + oW3, mom, e3T, H, F, stream);
    run_transpose(dW1 + odW1, d1T, F, H, stream);
    run_transpose(dW2 + oW3, d2T, H, F, stream);
    ema_kernel<<<(H + 255) / 256, 256, 0, stream>>>(eg1, tg1 + oV, g1 + oV, mom, H);
    ema_kernel<<<(H + 255) / 256, 256, 0, stream>>>(eb1, tb1 + oV, b1 + oV, mom, H);
    ema_kernel<<<(H + 255) / 256, 256, 0, stream>>>(eg2, tg2 + oV, g2 + oV, mom, H);
    ema_kernel<<<(H + 255) / 256, 256, 0, stream>>>(eb2, tb2 + oV, b2 + oV, mom, H);

    // online FCN -> z_q[v] (f32) + zqb raw (bf16, feeds MLP)
    run_gemm(xb, w1T, buf1, N, H, D, stream);
    run_bn_hidden(buf1, g1 + oV, b1 + oV, abuf, stats, N, H, stream);
    run_gemm(abuf, w2T, buf1, N, H, H, stream);
    run_bn_hidden(buf1, g2 + oV, b2 + oV, abuf, stats, N, H, stream);
    run_gemm(abuf, w3T, buf1, N, F, H, stream);
    run_bn_final(buf1, zq + oZ, zqb + oZ, stats, N, F, stream);

    // target FCN (EMA params) -> z_k[v]
    run_gemm(xb, e1T, buf1, N, H, D, stream);
    run_bn_hidden(buf1, eg1, eb1, abuf, stats, N, H, stream);
    run_gemm(abuf, e2T, buf1, N, H, H, stream);
    run_bn_hidden(buf1, eg2, eb2, abuf, stats, N, H, stream);
    run_gemm(abuf, e3T, buf1, N, F, H, stream);
    run_bn_final(buf1, zk + oZ, zkb + oZ, stats, N, F, stream);

    // predictor MLP on raw z_q -> p[v] (f32)
    run_gemm_bias_relu_bf16(zqb + oZ, d1T, db1 + oV, mlpb, N, H, F, stream);
    run_gemm_bias(mlpb, d2T, db2 + oFv, pp + oZ, N, F, H, stream);
  }

  // row L2-normalize (in place, f32), then bf16 copies for similarity GEMMs
  l2norm_rows<<<V * N, 256, 0, stream>>>(zq, F);
  l2norm_rows<<<V * N, 256, 0, stream>>>(zk, F);
  l2norm_rows<<<V * N, 256, 0, stream>>>(pp, F);
  cvt_bf16<<<(V * N * F + 255) / 256, 256, 0, stream>>>(zq, zqb, V * N * F);
  cvt_bf16<<<(V * N * F + 255) / 256, 256, 0, stream>>>(zk, zkb, V * N * F);
  cvt_bf16<<<(V * N * F + 255) / 256, 256, 0, stream>>>(pp, ppb, V * N * F);

  // losses: per key view kv, mp[kv] serves intra + two inter terms
  const float w_intra = 1.f / (3.f * (float)N * (float)N);
  const float w_inter = 1.f / (6.f * (float)N * (float)N);
  for (int kv = 0; kv < V; ++kv) {
    const bf16_t* kn = zkb + (size_t)kv * N * F;
    run_gemm(kn, kn, Gb, N, N, F, stream);              // zk_n @ zk_n^T
    affinity_rows<<<N, 256, 0, stream>>>(Gb, mpb, N);

    run_gemm(zqb + (size_t)kv * N * F, kn, Gb, N, N, F, stream);
    loss_rows<<<N, 256, 0, stream>>>(Gb, mpb, out, w_intra, N);

    for (int qv = 0; qv < V; ++qv) {
      if (qv == kv) continue;
      run_gemm(ppb + (size_t)qv * N * F, kn, Gb, N, N, F, stream);
      loss_rows<<<N, 256, 0, stream>>>(Gb, mpb, out, w_inter, N);
    }
  }
}
